// Net_30099130811154
// MI455X (gfx1250) — matrix operand order repaired and checked by reading, not compiled, not run on hardware
//
#include <hip/hip_runtime.h>
#include <hip/hip_bf16.h>

#define N_NODES 2048
#define N_EDGES 65536
#define EMBED   64
#define H       128
#define CFEAT   320      // EMBED + H + H
#define NH      20       // head channels
#define ABLD    48       // padded (a|b) columns: 20 + 20 -> 48 (mult of 16)
#define AMAX_INIT 0x007FFFFFu  // ordered-uint encoding of -inf

typedef _Float16 half8 __attribute__((ext_vector_type(8)));
typedef _Float16 v16h  __attribute__((ext_vector_type(16)));
typedef float    v8f   __attribute__((ext_vector_type(8)));

__device__ __forceinline__ float leaky(float x) { return x > 0.f ? x : 0.01f * x; }
__device__ __forceinline__ unsigned fenc(float f) {
  unsigned u = __float_as_uint(f);
  return (u & 0x80000000u) ? ~u : (u | 0x80000000u);
}
__device__ __forceinline__ float fdec(unsigned u) {
  unsigned v = (u & 0x80000000u) ? (u & 0x7fffffffu) : ~u;
  return __uint_as_float(v);
}

// ---------------- WMMA GEMM: D[M x Nc] = act(A[M x K] @ Bw[Nc x K]^T + bias) ----
// A, Bw row-major f16; one wave (32 lanes) per 16x16 output tile; K % 32 == 0.
__global__ void k_gemm_wmma(const _Float16* __restrict__ A,
                            const _Float16* __restrict__ Bw,
                            const float* __restrict__ bias,
                            float* __restrict__ D,
                            _Float16* __restrict__ Dh,
                            int Nc, int K, int act) {
  const int tilesN = Nc >> 4;
  const int tm = blockIdx.x / tilesN;
  const int tn = blockIdx.x % tilesN;
  const int lane = threadIdx.x;
  const int g = lane >> 4;     // lane group (0: lanes 0-15, 1: lanes 16-31)
  const int r = lane & 15;     // row-of-A / row-of-Bw within tile
  const _Float16* Arow = A  + (size_t)(tm * 16 + r) * K;
  const _Float16* Brow = Bw + (size_t)(tn * 16 + r) * K;
  v8f acc = {};
  for (int k0 = 0; k0 < K; k0 += 32) {
    // A layout: halves 0-7 -> K = k0+g*8.. ; halves 8-15 -> K = k0+16+g*8..
    half8 alo = *(const half8*)(Arow + k0 + g * 8);
    half8 ahi = *(const half8*)(Arow + k0 + 16 + g * 8);
    // B layout: halves 0-7 -> K = k0+g*8.. ; halves 8-15 -> K = k0+16+g*8..  (column of B = row of Bw)
    half8 blo = *(const half8*)(Brow + k0 + g * 8);
    half8 bhi = *(const half8*)(Brow + k0 + 16 + g * 8);
    v16h av, bv;
#pragma unroll
    for (int i = 0; i < 8; ++i) {
      av[i] = alo[i]; av[i + 8] = ahi[i];
      bv[i] = blo[i]; bv[i + 8] = bhi[i];
    }
    acc = __builtin_amdgcn_wmma_f32_16x16x32_f16(false, av, false, bv,
                                                 (short)0, acc, false, false);
  }
  const int n = tn * 16 + r;
  const float bn = bias ? bias[n] : 0.f;
#pragma unroll
  for (int i = 0; i < 8; ++i) {
    const int m = tm * 16 + g * 8 + i;       // C/D layout: VGPR i -> M = g*8+i
    float v = acc[i] + bn;
    if (act) v = fmaxf(v, 0.f);
    D[(size_t)m * Nc + n] = v;
    if (Dh) Dh[(size_t)m * Nc + n] = (_Float16)v;
  }
}

// ---------------- small prep kernels ------------------------------------------
__global__ void k_cvt_strided(const float* __restrict__ src, int ld, int cols,
                              _Float16* __restrict__ dst, int total) {
  int idx = blockIdx.x * blockDim.x + threadIdx.x;
  if (idx >= total) return;
  int row = idx / cols, c = idx % cols;
  dst[idx] = (_Float16)src[(size_t)row * ld + c];
}

__global__ void k_extract_col(const float* __restrict__ src, int ld, int col,
                              float* __restrict__ dst, int rows) {
  int r = blockIdx.x * blockDim.x + threadIdx.x;
  if (r < rows) dst[r] = src[(size_t)r * ld + col];
}

__global__ void k_build_wab(const float* __restrict__ W1, _Float16* __restrict__ wab) {
  int idx = blockIdx.x * blockDim.x + threadIdx.x;
  if (idx >= ABLD * CFEAT) return;
  int r = idx / CFEAT, c = idx % CFEAT;
  float v = 0.f;
  if (r < NH)            v = W1[(size_t)r * (2 * CFEAT) + c];
  else if (r < 2 * NH)   v = W1[(size_t)(r - NH) * (2 * CFEAT) + CFEAT + c];
  wab[idx] = (_Float16)v;
}

__global__ void k_embed(const int* __restrict__ ids, const float* __restrict__ emb,
                        float* __restrict__ x, _Float16* __restrict__ xh) {
  const int n = blockIdx.x, k = threadIdx.x;
  const float v = emb[(size_t)ids[n] * EMBED + k];
  x [(size_t)n * EMBED + k] = v;
  xh[(size_t)n * EMBED + k] = (_Float16)v;
}

// r[n] = X[n,:len] . w  (len <= 128)
__global__ void k_dot(const float* __restrict__ X, int len, const float* __restrict__ w,
                      float* __restrict__ r) {
  __shared__ float red[128];
  const int n = blockIdx.x, k = threadIdx.x;
  red[k] = (k < len) ? X[(size_t)n * len + k] * w[k] : 0.f;
  __syncthreads();
  for (int s = 64; s > 0; s >>= 1) { if (k < s) red[k] += red[k + s]; __syncthreads(); }
  if (k == 0) r[n] = red[0];
}

__global__ void k_init_seg(float* __restrict__ S, float* __restrict__ denom,
                           unsigned* __restrict__ amax) {
  const int idx = blockIdx.x * blockDim.x + threadIdx.x;
  if (idx < N_NODES * H) S[idx] = 0.f;
  if (idx < N_NODES) { denom[idx] = 0.f; amax[idx] = AMAX_INIT; }
}

// ---------------- edge passes --------------------------------------------------
// pass A: alpha_raw per edge + segment max over dst
__global__ void k_edgeA(const float* __restrict__ P, const float* __restrict__ wcol,
                        const float* __restrict__ attl, const float* __restrict__ rdst,
                        const int* __restrict__ src, const int* __restrict__ dst,
                        const float* __restrict__ eattr,
                        float* __restrict__ alpha, unsigned* __restrict__ amax) {
  __shared__ float red[2][H];
  const int k = threadIdx.x, ty = threadIdx.y;
  const int e = blockIdx.x * 2 + ty;
  const int s = src[e], d = dst[e];
  const float h = leaky(P[(size_t)s * H + k] + eattr[e] * wcol[k]);
  red[ty][k] = h * attl[k];
  __syncthreads();
  for (int st = H / 2; st > 0; st >>= 1) {
    if (k < st) red[ty][k] += red[ty][k + st];
    __syncthreads();
  }
  if (k == 0) {
    const float ar = leaky(red[ty][0] + rdst[d]);
    alpha[e] = ar;
    atomicMax(amax + d, fenc(ar));
  }
}

// pass B: S[dst] += w * h ; denom[dst] += w   (w = exp(alpha - amax[dst]))
__global__ void k_edgeB(const float* __restrict__ P, const float* __restrict__ wcol,
                        const int* __restrict__ src, const int* __restrict__ dst,
                        const float* __restrict__ eattr,
                        const float* __restrict__ alpha, const unsigned* __restrict__ amax,
                        float* __restrict__ S, float* __restrict__ denom) {
  const int k = threadIdx.x, ty = threadIdx.y;
  const int e = blockIdx.x * 2 + ty;
  const int s = src[e], d = dst[e];
  float am = fdec(amax[d]);
  if (!(am > -3.0e38f)) am = 0.f;          // reference: non-finite segment max -> 0
  const float w = __expf(alpha[e] - am);
  if (k == 0) atomicAdd(denom + d, w);
  const float h = leaky(P[(size_t)s * H + k] + eattr[e] * wcol[k]);
  atomicAdd(S + (size_t)d * H + k, w * h);
}

__global__ void k_nodeT(const float* __restrict__ S, const float* __restrict__ denom,
                        _Float16* __restrict__ Th) {
  const int n = blockIdx.x, k = threadIdx.x;
  Th[(size_t)n * H + k] = (_Float16)(S[(size_t)n * H + k] / (denom[n] + 1e-16f));
}

__global__ void k_feat(const float* __restrict__ x, const float* __restrict__ h1,
                       const float* __restrict__ h2, _Float16* __restrict__ fh) {
  const int n = blockIdx.x;
  for (int c = threadIdx.x; c < CFEAT; c += blockDim.x) {
    float v = (c < EMBED)      ? x [(size_t)n * EMBED + c]
            : (c < EMBED + H)  ? h1[(size_t)n * H + (c - EMBED)]
                               : h2[(size_t)n * H + (c - EMBED - H)];
    fh[(size_t)n * CFEAT + c] = (_Float16)v;
  }
}

// ---------------- pairwise head + row softmax ----------------------------------
__global__ void k_head(const float* __restrict__ ab, const float* __restrict__ msk,
                       const float* __restrict__ b1, const float* __restrict__ W2,
                       const float* __restrict__ b2, float* __restrict__ out) {
  __shared__ float arow[NH];
  __shared__ float logits[N_NODES];
  __shared__ float red[256];
  const int i = blockIdx.x, tid = threadIdx.x;
  if (tid < NH) arow[tid] = ab[(size_t)i * ABLD + tid] + b1[tid];
  __syncthreads();
  const float bb = b2[0];
  for (int j = tid; j < N_NODES; j += 256) {
    const float* bj = ab + (size_t)j * ABLD + NH;
    float s = bb;
#pragma unroll
    for (int c = 0; c < NH; ++c) s += fmaxf(arow[c] + bj[c], 0.f) * W2[c];
    logits[j] = s * msk[(size_t)i * N_NODES + j];
  }
  __syncthreads();
  float mx = -3.4e38f;
  for (int j = tid; j < N_NODES; j += 256) mx = fmaxf(mx, logits[j]);
  red[tid] = mx; __syncthreads();
  for (int s = 128; s > 0; s >>= 1) { if (tid < s) red[tid] = fmaxf(red[tid], red[tid + s]); __syncthreads(); }
  mx = red[0]; __syncthreads();
  float sm = 0.f;
  for (int j = tid; j < N_NODES; j += 256) { float e = __expf(logits[j] - mx); logits[j] = e; sm += e; }
  red[tid] = sm; __syncthreads();
  for (int s = 128; s > 0; s >>= 1) { if (tid < s) red[tid] += red[tid + s]; __syncthreads(); }
  const float inv = 1.f / red[0];
  for (int j = tid; j < N_NODES; j += 256) out[(size_t)i * N_NODES + j] = logits[j] * inv;
}

// ---------------- launcher -----------------------------------------------------
extern "C" void kernel_launch(void* const* d_in, const int* in_sizes, int n_in,
                              void* d_out, int out_size, void* d_ws, size_t ws_size,
                              hipStream_t stream) {
  (void)in_sizes; (void)n_in; (void)out_size; (void)ws_size;
  const int*   node_ids = (const int*)  d_in[0];
  const int*   ei       = (const int*)  d_in[1];   // [2,E]: src then dst
  const float* eattr    = (const float*)d_in[2];
  const float* msk      = (const float*)d_in[3];
  const float* emb      = (const float*)d_in[4];
  const float* l0_lin1  = (const float*)d_in[5];   // [128,65]
  const float* l0_lin2  = (const float*)d_in[6];   // [128,128]
  const float* l0_att_l = (const float*)d_in[7];
  const float* l0_att_r = (const float*)d_in[8];
  const float* l0_bias  = (const float*)d_in[9];
  const float* l1_lin1  = (const float*)d_in[10];  // [128,129]
  const float* l1_lin2  = (const float*)d_in[11];
  const float* l1_att_l = (const float*)d_in[12];
  const float* l1_att_r = (const float*)d_in[13];
  const float* l1_bias  = (const float*)d_in[14];
  const float* W1       = (const float*)d_in[15];  // [20,640]
  const float* b1       = (const float*)d_in[16];
  const float* W2       = (const float*)d_in[17];
  const float* b2       = (const float*)d_in[18];
  float* out = (float*)d_out;
  const int* esrc = ei;
  const int* edst = ei + N_EDGES;

  // workspace carve (bump allocator, 256B aligned)
  char* p = (char*)d_ws;
  auto alloc = [&](size_t nbytes) -> char* {
    char* q = p; p += (nbytes + 255) & ~(size_t)255; return q;
  };
  float*     x     = (float*)    alloc(sizeof(float) * N_NODES * EMBED);
  float*     P     = (float*)    alloc(sizeof(float) * N_NODES * H);
  float*     S     = (float*)    alloc(sizeof(float) * N_NODES * H);
  float*     denom = (float*)    alloc(sizeof(float) * N_NODES);
  unsigned*  amax  = (unsigned*) alloc(sizeof(unsigned) * N_NODES);
  float*     alpha = (float*)    alloc(sizeof(float) * N_EDGES);
  float*     rvec  = (float*)    alloc(sizeof(float) * N_NODES);
  float*     h1    = (float*)    alloc(sizeof(float) * N_NODES * H);
  float*     h2    = (float*)    alloc(sizeof(float) * N_NODES * H);
  float*     ab    = (float*)    alloc(sizeof(float) * N_NODES * ABLD);
  _Float16*  xh    = (_Float16*) alloc(sizeof(_Float16) * N_NODES * EMBED);
  _Float16*  Th    = (_Float16*) alloc(sizeof(_Float16) * N_NODES * H);
  _Float16*  h1h   = (_Float16*) alloc(sizeof(_Float16) * N_NODES * H);
  _Float16*  feath = (_Float16*) alloc(sizeof(_Float16) * N_NODES * CFEAT);
  _Float16*  w01   = (_Float16*) alloc(sizeof(_Float16) * H * EMBED);
  _Float16*  w02   = (_Float16*) alloc(sizeof(_Float16) * H * H);
  _Float16*  w11   = (_Float16*) alloc(sizeof(_Float16) * H * H);
  _Float16*  w12   = (_Float16*) alloc(sizeof(_Float16) * H * H);
  _Float16*  wab   = (_Float16*) alloc(sizeof(_Float16) * ABLD * CFEAT);
  float*     w0col = (float*)    alloc(sizeof(float) * H);
  float*     w1col = (float*)    alloc(sizeof(float) * H);

  // weight prep (f32 -> f16, strip edge-attr column)
  k_cvt_strided<<<(H * EMBED + 255) / 256, 256, 0, stream>>>(l0_lin1, EMBED + 1, EMBED, w01, H * EMBED);
  k_cvt_strided<<<(H * H + 255) / 256, 256, 0, stream>>>(l0_lin2, H, H, w02, H * H);
  k_cvt_strided<<<(H * H + 255) / 256, 256, 0, stream>>>(l1_lin1, H + 1, H, w11, H * H);
  k_cvt_strided<<<(H * H + 255) / 256, 256, 0, stream>>>(l1_lin2, H, H, w12, H * H);
  k_build_wab<<<(ABLD * CFEAT + 255) / 256, 256, 0, stream>>>(W1, wab);
  k_extract_col<<<1, H, 0, stream>>>(l0_lin1, EMBED + 1, EMBED, w0col, H);
  k_extract_col<<<1, H, 0, stream>>>(l1_lin1, H + 1, H, w1col, H);

  const int gemmNH   = (N_NODES / 16) * (H / 16);     // 1024 tiles
  const int gemmNAB  = (N_NODES / 16) * (ABLD / 16);  // 384 tiles
  const dim3 eblk(H, 2);
  const int egrid = N_EDGES / 2;

  // ---- embed + layer 0 ----
  k_embed<<<N_NODES, EMBED, 0, stream>>>(node_ids, emb, x, xh);
  k_dot<<<N_NODES, 128, 0, stream>>>(x, EMBED, l0_att_r, rvec);
  k_gemm_wmma<<<gemmNH, 32, 0, stream>>>(xh, w01, nullptr, P, nullptr, H, EMBED, 0);
  k_init_seg<<<(N_NODES * H + 255) / 256, 256, 0, stream>>>(S, denom, amax);
  k_edgeA<<<egrid, eblk, 0, stream>>>(P, w0col, l0_att_l, rvec, esrc, edst, eattr, alpha, amax);
  k_edgeB<<<egrid, eblk, 0, stream>>>(P, w0col, esrc, edst, eattr, alpha, amax, S, denom);
  k_nodeT<<<N_NODES, H, 0, stream>>>(S, denom, Th);
  k_gemm_wmma<<<gemmNH, 32, 0, stream>>>(Th, w02, l0_bias, h1, h1h, H, H, 1);

  // ---- layer 1 ----
  k_dot<<<N_NODES, 128, 0, stream>>>(h1, H, l1_att_r, rvec);
  k_gemm_wmma<<<gemmNH, 32, 0, stream>>>(h1h, w11, nullptr, P, nullptr, H, H, 0);
  k_init_seg<<<(N_NODES * H + 255) / 256, 256, 0, stream>>>(S, denom, amax);
  k_edgeA<<<egrid, eblk, 0, stream>>>(P, w1col, l1_att_l, rvec, esrc, edst, eattr, alpha, amax);
  k_edgeB<<<egrid, eblk, 0, stream>>>(P, w1col, esrc, edst, eattr, alpha, amax, S, denom);
  k_nodeT<<<N_NODES, H, 0, stream>>>(S, denom, Th);
  k_gemm_wmma<<<gemmNH, 32, 0, stream>>>(Th, w12, l1_bias, h2, nullptr, H, H, 1);

  // ---- pairwise head ----
  k_feat<<<N_NODES, 128, 0, stream>>>(x, h1, h2, feath);
  k_gemm_wmma<<<gemmNAB, 32, 0, stream>>>(feath, wab, nullptr, ab, nullptr, ABLD, CFEAT, 0);
  k_head<<<N_NODES, 256, 0, stream>>>(ab, msk, b1, W2, b2, out);
}
